// EdgeNetwork_24137716203976
// MI455X (gfx1250) — matrix operand order repaired and checked by reading, not compiled, not run on hardware
//
#include <hip/hip_runtime.h>
#include <hip/hip_bf16.h>

typedef __attribute__((ext_vector_type(16))) _Float16 v16h;
typedef __attribute__((ext_vector_type(8)))  float    v8f;
typedef __attribute__((ext_vector_type(2)))  _Float16 h2;

#define F_FEAT 16
#define F_H    64
#define HID    128
#define OUTD   64
#define IN_DIM 162
#define KPAD   192            // 162 padded up to 6 * 32
#define ROWS   256            // edges per tile (8 waves * 32 rows)
#define XS     200            // x-tile row stride (halves): 192 + 8 pad (bank spread)
#define HS     136            // h-tile row stride (halves): 128 + 8 pad
#define NFRAG_W1 48           // 6 k-steps * 8 n-tiles
#define NFRAG_W2 32           // 4 * 8
#define NFRAG_W3 16           // 4 * 4
#define NFRAG_TOT 96
#define FRAG_HALVES 512       // 16x16x32-f16 B fragment: 32 lanes * 16 halves
#define WFRAG_HALVES (NFRAG_TOT * FRAG_HALVES)
#define SMEM_HALVES (WFRAG_HALVES + ROWS * XS + ROWS * HS)
#define SMEM_BYTES  (SMEM_HALVES * 2)   // 270336 B < 320 KB WGP LDS

// ---------------------------------------------------------------------------
// Prep: swizzle f32 weights into f16 WMMA B-fragment order in d_ws.
// B 32x16 f16 layout (CDNA5 ISA): lanes 0-15 -> N=lane, K=(h&7)+16*(h>=8);
// lanes 16-31 -> N=lane-16, K=8+(h&7)+16*(h>=8). frag[f][lane][h] at f*512 + lane*16 + h.
// ---------------------------------------------------------------------------
__global__ void prep_weights(const float* __restrict__ W1,
                             const float* __restrict__ W2,
                             const float* __restrict__ W3,
                             _Float16* __restrict__ frag) {
  int idx = blockIdx.x * 256 + threadIdx.x;
  if (idx >= WFRAG_HALVES) return;
  int f    = idx >> 9;
  int r    = idx & 511;
  int lane = r >> 4;
  int h    = r & 15;
  int nloc = lane & 15;
  int kloc = (h & 7) + ((h >= 8) ? 16 : 0) + ((lane >= 16) ? 8 : 0);
  float v;
  if (f < NFRAG_W1) {
    int ks = f >> 3, nt = f & 7;
    int K = ks * 32 + kloc, N = nt * 16 + nloc;
    v = (K < IN_DIM) ? W1[K * HID + N] : 0.f;        // zero-pad K 162..191
  } else if (f < NFRAG_W1 + NFRAG_W2) {
    int g = f - NFRAG_W1;
    int ks = g >> 3, nt = g & 7;
    int K = ks * 32 + kloc, N = nt * 16 + nloc;
    v = W2[K * HID + N];
  } else {
    int g = f - (NFRAG_W1 + NFRAG_W2);
    int ks = g >> 2, nt = g & 3;
    int K = ks * 32 + kloc, N = nt * 16 + nloc;
    v = W3[K * OUTD + N];
  }
  frag[idx] = (_Float16)v;
}

// ---------------------------------------------------------------------------
// Fragment loaders (A 16x32 f16 layout per ISA: lanes 0-15 M=lane, K runs
// {koff..koff+7, koff+16..koff+23} with koff = 8*(lane>=16)).
// ---------------------------------------------------------------------------
__device__ __forceinline__ v16h ld_afrag(const _Float16* p) {
  union { uint4 u[2]; v16h v; } f;
  f.u[0] = *(const uint4*)(p);
  f.u[1] = *(const uint4*)(p + 16);
  return f.v;
}
__device__ __forceinline__ v16h ld_bfrag(const _Float16* p) {
  union { uint4 u[2]; v16h v; } f;
  f.u[0] = *(const uint4*)(p);
  f.u[1] = *(const uint4*)(p + 8);
  return f.v;
}
__device__ __forceinline__ void st2(_Float16* p, float a, float b) {
  h2 t; t.x = (_Float16)a; t.y = (_Float16)b;
  *(h2*)p = t;
}
__device__ __forceinline__ v8f wmma_f16(v16h a, v16h b, v8f c) {
  return __builtin_amdgcn_wmma_f32_16x16x32_f16(false, a, false, b, (short)0, c,
                                                false, false);
}

// ---------------------------------------------------------------------------
// Main fused kernel: 256 threads = 8 wave32; each wave owns 32 rows (2 A
// fragments) for staging AND GEMM; B fragments are software-pipelined one
// step ahead so WMMA pairs overlap the next fragment's LDS latency.
// ---------------------------------------------------------------------------
__global__ void __launch_bounds__(256, 1)
edge_mlp_kernel(const float* __restrict__ nodef, const float* __restrict__ nodeh,
                const float* __restrict__ dist,
                const int* __restrict__ src_idx, const int* __restrict__ dst_idx,
                const _Float16* __restrict__ wfrag_g,
                const float* __restrict__ b1, const float* __restrict__ b2,
                const float* __restrict__ b3,
                float* __restrict__ out_attn, float* __restrict__ out_srch,
                float* __restrict__ out_edgeh,
                int E, int ntiles) {
  extern __shared__ _Float16 smem[];
  _Float16* wfrag = smem;                        // 96 KB of B fragments
  _Float16* xtile = smem + WFRAG_HALVES;         // 256 x 200 halves
  _Float16* htile = xtile + ROWS * XS;           // 256 x 136 halves

  const int tid = threadIdx.x;

  // cooperative weight-fragment load: global ws -> LDS (uint4 = 8 halves)
  {
    const uint4* src = (const uint4*)wfrag_g;
    uint4*       dst = (uint4*)wfrag;
    for (int i = tid; i < WFRAG_HALVES / 8; i += 256) dst[i] = src[i];
  }
  __syncthreads();   // only workgroup-wide barrier in the kernel

  const int lane = tid & 31;
  const int wave = tid >> 5;
  const int wrow = wave * 32;              // this wave's 32 GEMM rows
  const int moff = (lane >= 16) ? 8 : 0;   // C/D layout: M = vgpr + moff
  const int ncol = lane & 15;              // C/D layout: N = ntile*16 + ncol
  const int aoff = (lane >> 4) << 3;       // A layout K sub-offset (0 or 8)
  const _Float16* bbase = wfrag + lane * 16;

  h2 zz; zz.x = (_Float16)0.f; zz.y = (_Float16)0.f;
  const v8f vzero = {0.f, 0.f, 0.f, 0.f, 0.f, 0.f, 0.f, 0.f};

  for (int tile = blockIdx.x; tile < ntiles; tile += gridDim.x) {
    const int ebase = tile * ROWS;

    // ------- stage x row (1 thread per edge; rows are wave-local) ----------
    {
      _Float16* xrow = xtile + tid * XS;
      const int e = ebase + tid;
      if (e < E) {
        const int d = dst_idx[e];
        const int s = src_idx[e];
        const float4* df = (const float4*)(nodef + (size_t)d * F_FEAT);
        const float4* sf = (const float4*)(nodef + (size_t)s * F_FEAT);
#pragma unroll
        for (int i = 0; i < 4; ++i) {
          float4 v = df[i];
          st2(xrow + i * 4 + 0, v.x, v.y);
          st2(xrow + i * 4 + 2, v.z, v.w);
          float4 w = sf[i];
          st2(xrow + 80 + i * 4 + 0, w.x, w.y);
          st2(xrow + 80 + i * 4 + 2, w.z, w.w);
        }
        // dst_h / src_h: stage f16, copy src_h out in f32, f32 attention dot
        const float4* dh = (const float4*)(nodeh + (size_t)d * F_H);
        const float4* sh = (const float4*)(nodeh + (size_t)s * F_H);
        float4* o4 = (float4*)(out_srch + (size_t)e * F_H);
        float dot = 0.f;
#pragma unroll
        for (int i = 0; i < 16; ++i) {
          float4 a = dh[i];
          float4 b = sh[i];
          st2(xrow + 16 + i * 4 + 0, a.x, a.y);
          st2(xrow + 16 + i * 4 + 2, a.z, a.w);
          st2(xrow + 96 + i * 4 + 0, b.x, b.y);
          st2(xrow + 96 + i * 4 + 2, b.z, b.w);
          o4[i] = b;
          dot = fmaf(a.x, b.x, dot);
          dot = fmaf(a.y, b.y, dot);
          dot = fmaf(a.z, b.z, dot);
          dot = fmaf(a.w, b.w, dot);
        }
        const float attn = 1.f / (1.f + __expf(-dot * 0.125f)); // 1/sqrt(64)
        out_attn[e] = attn;
        st2(xrow + 160, dist[e], attn);
#pragma unroll
        for (int i = 0; i < 15; ++i) *(h2*)(xrow + 162 + 2 * i) = zz; // pad 162..191
      } else {
#pragma unroll
        for (int i = 0; i < 96; ++i) *(h2*)(xrow + 2 * i) = zz;
      }
    }
    // no barrier: staged rows are wave-local; LDS is in-order per wave.

    // ---------------- layer 1: x[32x192] @ W1[192x128] ---------------------
    v8f acc0[8], acc1[8];
#pragma unroll
    for (int nt = 0; nt < 8; ++nt) { acc0[nt] = vzero; acc1[nt] = vzero; }
    {
      const _Float16* arow0 = xtile + (wrow + ncol) * XS + aoff;
      const _Float16* arow1 = arow0 + 16 * XS;
      v16h b = ld_bfrag(bbase);                       // pipeline prologue
#pragma unroll
      for (int ks = 0; ks < 6; ++ks) {
        const v16h a0 = ld_afrag(arow0 + ks * 32);
        const v16h a1 = ld_afrag(arow1 + ks * 32);
#pragma unroll
        for (int nt = 0; nt < 8; ++nt) {
          const int nx = ks * 8 + nt + 1;
          v16h bn = b;
          if (nx < NFRAG_W1) bn = ld_bfrag(bbase + nx * FRAG_HALVES);
          acc0[nt] = wmma_f16(a0, b, acc0[nt]);
          acc1[nt] = wmma_f16(a1, b, acc1[nt]);
          b = bn;
        }
      }
    }
    // bias + ReLU -> h tile (f16)
#pragma unroll
    for (int nt = 0; nt < 8; ++nt) {
      const int N = nt * 16 + ncol;
      const float bb = b1[N];
#pragma unroll
      for (int v = 0; v < 8; ++v) {
        const float f0 = acc0[nt][v] + bb;
        const float f1 = acc1[nt][v] + bb;
        htile[(wrow + moff + v) * HS + N]      = (_Float16)(f0 > 0.f ? f0 : 0.f);
        htile[(wrow + 16 + moff + v) * HS + N] = (_Float16)(f1 > 0.f ? f1 : 0.f);
      }
    }

    // ---------------- layer 2: h1[32x128] @ W2[128x128] --------------------
#pragma unroll
    for (int nt = 0; nt < 8; ++nt) { acc0[nt] = vzero; acc1[nt] = vzero; }
    {
      const _Float16* arow0 = htile + (wrow + ncol) * HS + aoff;
      const _Float16* arow1 = arow0 + 16 * HS;
      v16h b = ld_bfrag(bbase + NFRAG_W1 * FRAG_HALVES);
#pragma unroll
      for (int ks = 0; ks < 4; ++ks) {
        const v16h a0 = ld_afrag(arow0 + ks * 32);
        const v16h a1 = ld_afrag(arow1 + ks * 32);
#pragma unroll
        for (int nt = 0; nt < 8; ++nt) {
          const int nx = ks * 8 + nt + 1;
          v16h bn = b;
          if (nx < NFRAG_W2)
            bn = ld_bfrag(bbase + (NFRAG_W1 + nx) * FRAG_HALVES);
          acc0[nt] = wmma_f16(a0, b, acc0[nt]);
          acc1[nt] = wmma_f16(a1, b, acc1[nt]);
          b = bn;
        }
      }
    }
#pragma unroll
    for (int nt = 0; nt < 8; ++nt) {
      const int N = nt * 16 + ncol;
      const float bb = b2[N];
#pragma unroll
      for (int v = 0; v < 8; ++v) {
        const float f0 = acc0[nt][v] + bb;
        const float f1 = acc1[nt][v] + bb;
        htile[(wrow + moff + v) * HS + N]      = (_Float16)(f0 > 0.f ? f0 : 0.f);
        htile[(wrow + 16 + moff + v) * HS + N] = (_Float16)(f1 > 0.f ? f1 : 0.f);
      }
    }

    // ---------------- layer 3: h2[32x128] @ W3[128x64] ---------------------
    v8f acc30[4], acc31[4];
#pragma unroll
    for (int nt = 0; nt < 4; ++nt) { acc30[nt] = vzero; acc31[nt] = vzero; }
    {
      const _Float16* arow0 = htile + (wrow + ncol) * HS + aoff;
      const _Float16* arow1 = arow0 + 16 * HS;
      v16h b = ld_bfrag(bbase + (NFRAG_W1 + NFRAG_W2) * FRAG_HALVES);
#pragma unroll
      for (int ks = 0; ks < 4; ++ks) {
        const v16h a0 = ld_afrag(arow0 + ks * 32);
        const v16h a1 = ld_afrag(arow1 + ks * 32);
#pragma unroll
        for (int nt = 0; nt < 4; ++nt) {
          const int nx = ks * 4 + nt + 1;
          v16h bn = b;
          if (nx < NFRAG_W3)
            bn = ld_bfrag(bbase + (NFRAG_W1 + NFRAG_W2 + nx) * FRAG_HALVES);
          acc30[nt] = wmma_f16(a0, b, acc30[nt]);
          acc31[nt] = wmma_f16(a1, b, acc31[nt]);
          b = bn;
        }
      }
    }
    // bias + f32 store to edge_h (lanes 0-15 / 16-31 each cover a 64B row run)
#pragma unroll
    for (int nt = 0; nt < 4; ++nt) {
      const int N = nt * 16 + ncol;
      const float bb = b3[N];
#pragma unroll
      for (int v = 0; v < 8; ++v) {
        const int e0 = ebase + wrow + moff + v;
        const int e1 = e0 + 16;
        if (e0 < E) out_edgeh[(size_t)e0 * OUTD + N] = acc30[nt][v] + bb;
        if (e1 < E) out_edgeh[(size_t)e1 * OUTD + N] = acc31[nt][v] + bb;
      }
    }
  }
}

// ---------------------------------------------------------------------------
extern "C" void kernel_launch(void* const* d_in, const int* in_sizes, int n_in,
                              void* d_out, int out_size, void* d_ws, size_t ws_size,
                              hipStream_t stream) {
  const float* nodef = (const float*)d_in[0];
  const float* nodeh = (const float*)d_in[1];
  const float* dist  = (const float*)d_in[2];
  const int*   srci  = (const int*)d_in[3];
  const int*   dsti  = (const int*)d_in[4];
  const float* W1 = (const float*)d_in[5];
  const float* b1 = (const float*)d_in[6];
  const float* W2 = (const float*)d_in[7];
  const float* b2 = (const float*)d_in[8];
  const float* W3 = (const float*)d_in[9];
  const float* b3 = (const float*)d_in[10];

  const int E = in_sizes[2];                 // n_edges (distance length)
  _Float16* wfrag_g = (_Float16*)d_ws;       // 96 KB of swizzled f16 weights

  float* out       = (float*)d_out;          // [attn E | src_h E*64 | edge_h E*64]
  float* out_attn  = out;
  float* out_srch  = out + E;
  float* out_edgeh = out + E + (size_t)E * OUTD;

  prep_weights<<<(WFRAG_HALVES + 255) / 256, 256, 0, stream>>>(W1, W2, W3, wfrag_g);

  const int ntiles = (E + ROWS - 1) / ROWS;  // 6250 for E = 1.6M
  const int nblocks = ntiles < 1536 ? ntiles : 1536;

  (void)hipFuncSetAttribute((const void*)edge_mlp_kernel,
                            hipFuncAttributeMaxDynamicSharedMemorySize, SMEM_BYTES);
  edge_mlp_kernel<<<nblocks, 256, SMEM_BYTES, stream>>>(
      nodef, nodeh, dist, srci, dsti, wfrag_g, b1, b2, b3,
      out_attn, out_srch, out_edgeh, E, ntiles);
}
